// PairwiseVectorNet_12953621364813
// MI455X (gfx1250) — matrix operand order repaired and checked by reading, not compiled, not run on hardware
//
#include <hip/hip_runtime.h>

// ---------------------------------------------------------------------------
// PairwiseVectorNet fused kernel for gfx1250 (MI455X, wave32, WMMA)
//
//   vecs  = coords[:, J] - coords[:, I]            (B, 2016, 3)
//   d     = |vecs|                                  (B, 2016)
//   h1    = silu(d  @ W1 + b1)                      (B, 128)
//   h2    = silu(h1 @ W2 + b2)                      (B, 128)
//   s     = h2 @ W3 + b3                            (B, 2016)
//   mu    = sum_p s[b,p] * vecs[b,p,:]              (B, 3)
//
// GEMMs run as v_wmma_f32_16x16x32_bf16 with fp32 accumulation.
// Weights are converted once per launch into bf16, zero-padded to K=2048,
// and pre-swizzled into the 32x16 B-fragment lane layout in d_ws.
// ---------------------------------------------------------------------------

typedef __attribute__((ext_vector_type(16))) __bf16        v16bf;
typedef __attribute__((ext_vector_type(8)))  float         v8f;
typedef __attribute__((ext_vector_type(4)))  unsigned int  u32x4;

union FragU { u32x4 u[2]; v16bf v; };

#define N_ATOMS   64
#define N_PAIRS   2016
#define KPAD      2048
#define HIDDEN    128
#define BATCH     16384
#define BTILE     16

// ws layout (bf16 elements):
#define W1F_OFF   0                       // 64 ktiles * 8 ntiles * 512
#define W1F_SZ    (64 * 8 * 512)          // 262144
#define W2F_OFF   (W1F_OFF + W1F_SZ)
#define W2F_SZ    (4 * 8 * 512)           // 16384
#define W3F_OFF   (W2F_OFF + W2F_SZ)
#define W3F_SZ    (4 * 126 * 512)         // 258048
#define WF_TOTAL  (W3F_OFF + W3F_SZ)      // 536576 elements (~1.05 MB)

__device__ __forceinline__ unsigned short f2bf(float f) {
  unsigned int x = __float_as_uint(f);
  unsigned int r = x + 0x7FFFu + ((x >> 16) & 1u);   // round-to-nearest-even
  return (unsigned short)(r >> 16);
}

// silu via fast v_rcp_f32 (single trans op, ~1 ulp) instead of IEEE divide
__device__ __forceinline__ float silu(float x) {
  return x * __builtin_amdgcn_rcpf(1.0f + __expf(-x));
}

// ---------------------------------------------------------------------------
// Prep: convert W1/W2/W3 (fp32 row-major, K x N) into bf16 fragments.
// Fragment f = nt*KT + kt holds the 32x16 B tile in WMMA lane order:
//   element e = lane*16 + v*2 + h  ->  n = nt*16 + (lane&15)
//                                      k = kt*32 + (lane>>4)*16 + 2v + h
// ---------------------------------------------------------------------------
__global__ void convert_weights_kernel(const float* __restrict__ W1,
                                       const float* __restrict__ W2,
                                       const float* __restrict__ W3,
                                       unsigned short* __restrict__ out) {
  int t = blockIdx.x * blockDim.x + threadIdx.x;
  if (t >= WF_TOTAL) return;

  const float* W;
  int KT, Kreal, N, idx;
  if (t < W2F_OFF)      { W = W1; KT = 64; Kreal = N_PAIRS; N = HIDDEN;  idx = t - W1F_OFF; }
  else if (t < W3F_OFF) { W = W2; KT = 4;  Kreal = HIDDEN;  N = HIDDEN;  idx = t - W2F_OFF; }
  else                  { W = W3; KT = 4;  Kreal = HIDDEN;  N = N_PAIRS; idx = t - W3F_OFF; }

  int f    = idx >> 9;          // fragment index
  int e    = idx & 511;         // element within fragment
  int nt   = f / KT;
  int kt   = f - nt * KT;
  int lane = e >> 4;
  int v    = (e >> 1) & 7;
  int h    = e & 1;
  int n    = nt * 16 + (lane & 15);
  int k    = kt * 32 + (lane >> 4) * 16 + 2 * v + h;

  out[t] = (k < Kreal) ? f2bf(W[(long)k * N + n]) : (unsigned short)0;
}

// ---------------------------------------------------------------------------
// Main fused kernel: one block = 16 batch rows, 8 waves.
// ---------------------------------------------------------------------------
__global__ __launch_bounds__(256) void
pairnet_kernel(const float* __restrict__ coords,
               const unsigned short* __restrict__ wf,
               const float* __restrict__ b1,
               const float* __restrict__ b2,
               const float* __restrict__ b3,
               float* __restrict__ muOut,
               float* __restrict__ sOut) {
  const int tid  = threadIdx.x;
  const int lid  = tid & 31;
  const int wv   = tid >> 5;       // wave 0..7
  const int m16  = lid & 15;
  const int half = lid >> 4;
  const int b0   = blockIdx.x * BTILE;

  __shared__ __align__(16) float          coords_s[BTILE * N_ATOMS * 3]; // 12 KB
  __shared__ __align__(16) unsigned short dists_s [BTILE * KPAD];        // 64 KB
  __shared__ __align__(16) unsigned short h1_s    [BTILE * HIDDEN];      // 4 KB
  __shared__ __align__(16) unsigned short h2_s    [BTILE * HIDDEN];      // 4 KB
  __shared__               int            idx_s   [N_PAIRS];             // 8 KB
  __shared__               float          mu_s    [BTILE * 3];

  const unsigned short* w1f = wf + W1F_OFF;
  const unsigned short* w2f = wf + W2F_OFF;
  const unsigned short* w3f = wf + W3F_OFF;

  // ---- Phase A: load coords tile, zero mu accumulators -------------------
  for (int t = tid; t < BTILE * N_ATOMS * 3; t += 256)
    coords_s[t] = coords[(long)b0 * N_ATOMS * 3 + t];
  if (tid < BTILE * 3) mu_s[tid] = 0.0f;

  // ---- Phase B: build triu(k=1) pair index table -------------------------
  for (int p = tid; p < N_PAIRS; p += 256) {
    int i = 0, rem = p;
    while (rem >= (N_ATOMS - 1) - i) { rem -= (N_ATOMS - 1) - i; ++i; }
    int j = i + 1 + rem;
    idx_s[p] = (i << 8) | j;
  }
  __syncthreads();

  // ---- Phase C: pairwise distances -> bf16 LDS (zero-padded K) -----------
  for (int p = tid; p < KPAD; p += 256) {
    if (p < N_PAIRS) {
      int ij = idx_s[p];
      int i = ij >> 8, j = ij & 255;
      for (int bl = 0; bl < BTILE; ++bl) {
        const float* ci = &coords_s[bl * 192 + i * 3];
        const float* cj = &coords_s[bl * 192 + j * 3];
        float dx = cj[0] - ci[0], dy = cj[1] - ci[1], dz = cj[2] - ci[2];
        dists_s[bl * KPAD + p] = f2bf(__fsqrt_rn(dx * dx + dy * dy + dz * dz));
      }
    } else {
      for (int bl = 0; bl < BTILE; ++bl) dists_s[bl * KPAD + p] = 0;
    }
  }
  __syncthreads();

  // A fragment (16x32 bf16) from LDS, contiguous K layout matching B:
  //   lanes 0-15: K k0+0..15; lanes 16-31: K k0+16..31
  auto loadA = [&](const unsigned short* arr, int stride, int k0) -> v16bf {
    FragU u;
    u.u[0] = *(const u32x4*)&arr[m16 * stride + k0 + half * 16];
    u.u[1] = *(const u32x4*)&arr[m16 * stride + k0 + half * 16 + 8];
    return u.v;
  };
  // B fragment (32x16 bf16) pre-swizzled in global: 32 contiguous bytes/lane
  auto loadB = [&](const unsigned short* g, int fragElemBase) -> v16bf {
    FragU u;
    const u32x4* p = (const u32x4*)&g[fragElemBase + lid * 16];
    u.u[0] = p[0];
    u.u[1] = p[1];
    return u.v;
  };

  // ---- Phase D: GEMM1  (16 x 2048) @ (2048 x 128), wave wv owns N tile ---
  {
    v8f acc = {};
#pragma unroll 4
    for (int kt = 0; kt < KPAD / 32; ++kt) {
      v16bf a = loadA(dists_s, KPAD, kt * 32);
      v16bf b = loadB(w1f, (wv * 64 + kt) * 512);
      acc = __builtin_amdgcn_wmma_f32_16x16x32_bf16(false, a, false, b,
                                                    (short)0, acc, false, false);
    }
    float bv = b1[wv * 16 + m16];
#pragma unroll
    for (int r = 0; r < 8; ++r) {
      float y = silu(acc[r] + bv);
      h1_s[(r + 8 * half) * HIDDEN + wv * 16 + m16] = f2bf(y);
    }
  }
  __syncthreads();

  // ---- Phase E: GEMM2  (16 x 128) @ (128 x 128) --------------------------
  {
    v8f acc = {};
#pragma unroll
    for (int kt = 0; kt < 4; ++kt) {
      v16bf a = loadA(h1_s, HIDDEN, kt * 32);
      v16bf b = loadB(w2f, (wv * 4 + kt) * 512);
      acc = __builtin_amdgcn_wmma_f32_16x16x32_bf16(false, a, false, b,
                                                    (short)0, acc, false, false);
    }
    float bv = b2[wv * 16 + m16];
#pragma unroll
    for (int r = 0; r < 8; ++r) {
      float y = silu(acc[r] + bv);
      h2_s[(r + 8 * half) * HIDDEN + wv * 16 + m16] = f2bf(y);
    }
  }
  __syncthreads();

  // ---- Phase F: GEMM3 (16 x 128) @ (128 x 2016), s out + mu accumulate ---
  {
    v16bf aF[4];
#pragma unroll
    for (int kt = 0; kt < 4; ++kt) aF[kt] = loadA(h2_s, HIDDEN, kt * 32);

    float mux[8], muy[8], muz[8];
#pragma unroll
    for (int r = 0; r < 8; ++r) { mux[r] = 0.f; muy[r] = 0.f; muz[r] = 0.f; }

    for (int t = 0; t < 16; ++t) {
      int nt = wv + 8 * t;                 // wave-uniform
      if (nt >= N_PAIRS / 16) break;
      v8f acc = {};
#pragma unroll
      for (int kt = 0; kt < 4; ++kt) {
        v16bf b = loadB(w3f, (nt * 4 + kt) * 512);
        acc = __builtin_amdgcn_wmma_f32_16x16x32_bf16(false, aF[kt], false, b,
                                                      (short)0, acc, false, false);
      }
      int   p   = nt * 16 + m16;
      float b3v = b3[p];
      int   ij  = idx_s[p];
      int   i   = ij >> 8, j = ij & 255;
#pragma unroll
      for (int r = 0; r < 8; ++r) {
        int   M  = r + 8 * half;
        float sv = acc[r] + b3v;
        sOut[(long)(b0 + M) * N_PAIRS + p] = sv;
        const float* ci = &coords_s[M * 192 + i * 3];
        const float* cj = &coords_s[M * 192 + j * 3];
        mux[r] += sv * (cj[0] - ci[0]);
        muy[r] += sv * (cj[1] - ci[1]);
        muz[r] += sv * (cj[2] - ci[2]);
      }
    }
#pragma unroll
    for (int r = 0; r < 8; ++r) {
      int M = r + 8 * half;
      atomicAdd(&mu_s[M * 3 + 0], mux[r]);
      atomicAdd(&mu_s[M * 3 + 1], muy[r]);
      atomicAdd(&mu_s[M * 3 + 2], muz[r]);
    }
  }
  __syncthreads();

  // ---- Phase G: write mu -------------------------------------------------
  if (tid < BTILE * 3) muOut[(long)b0 * 3 + tid] = mu_s[tid];
}

extern "C" void kernel_launch(void* const* d_in, const int* in_sizes, int n_in,
                              void* d_out, int out_size, void* d_ws, size_t ws_size,
                              hipStream_t stream) {
  const float* coords = (const float*)d_in[0];
  const float* W1     = (const float*)d_in[1];
  const float* b1     = (const float*)d_in[2];
  const float* W2     = (const float*)d_in[3];
  const float* b2     = (const float*)d_in[4];
  const float* W3     = (const float*)d_in[5];
  const float* b3     = (const float*)d_in[6];

  unsigned short* wf = (unsigned short*)d_ws;
  float* muOut = (float*)d_out;
  float* sOut  = muOut + (long)BATCH * 3;

  int batch = in_sizes[0] / (N_ATOMS * 3);   // 16384

  convert_weights_kernel<<<(WF_TOTAL + 255) / 256, 256, 0, stream>>>(W1, W2, W3, wf);
  pairnet_kernel<<<batch / BTILE, 256, 0, stream>>>(coords, wf, b1, b2, b3, muOut, sOut);
}
